// TTTHead_45432164057777
// MI455X (gfx1250) — compile-verified, matrix operand order checked by reading, not run on hardware
//
#include <hip/hip_runtime.h>

// ---- CDNA5 WMMA types / helpers -------------------------------------------
typedef __attribute__((ext_vector_type(2)))  float  v2f;
typedef __attribute__((ext_vector_type(8)))  float  v8f;
typedef __attribute__((ext_vector_type(16))) __bf16 v16bf;

__device__ __forceinline__ v8f zero8() {
    v8f z;
#pragma unroll
    for (int i = 0; i < 8; ++i) z[i] = 0.f;
    return z;
}

// D = A(16x4 f32) * B(4x16 f32) + C   (exact-path GEMMs)
__device__ __forceinline__ v8f wmma_f32x4(v2f a, v2f b, v8f c) {
    return __builtin_amdgcn_wmma_f32_16x16x4_f32(false, a, false, b, (short)0, c, false, false);
}
// D = A(16x32 bf16) * B(32x16 bf16) + C   (dominant Wlr GEMM)
__device__ __forceinline__ v8f wmma_bf16(v16bf a, v16bf b, v8f c) {
    return __builtin_amdgcn_wmma_f32_16x16x32_bf16(false, a, false, b, (short)0, c, false, false);
}

__device__ __forceinline__ float sigmoidf(float x) { return 1.f / (1.f + __expf(-x)); }

#define N_  2048
#define D_  1024
#define F_  256
#define FF_ 65536
#define P_  65792   // F*F + F

// ---- K1: src -> bf16, zero post -------------------------------------------
__global__ void k_prep(const float* __restrict__ src, __bf16* __restrict__ srcb,
                       float* __restrict__ post) {
    int t = blockIdx.x * 256 + threadIdx.x;      // 524288 threads
    int base = t * 4;
    float4 f = *(const float4*)(src + base);
    union { __bf16 h[4]; uint2 u; } pk;
    pk.h[0] = (__bf16)f.x; pk.h[1] = (__bf16)f.y;
    pk.h[2] = (__bf16)f.z; pk.h[3] = (__bf16)f.w;
    *(uint2*)(srcb + base) = pk.u;
    if (t < P_) post[t] = 0.f;
}

// ---- K2: three projections  C = src @ theta  (f32 WMMA, K=1024) -----------
__global__ void k_proj(const float* __restrict__ src,
                       const float* __restrict__ tk, const float* __restrict__ tq,
                       const float* __restrict__ tv,
                       float* __restrict__ Tv, float* __restrict__ Qv, float* __restrict__ Lv) {
    const float* B; float* C;
    if      (blockIdx.y == 0) { B = tk; C = Tv; }
    else if (blockIdx.y == 1) { B = tq; C = Qv; }
    else                      { B = tv; C = Lv; }
    int wid = threadIdx.x >> 5, lane = threadIdx.x & 31, l16 = lane & 15, half = lane >> 4;
    int task = blockIdx.x * 8 + wid;             // 512 tasks
    int mtile = task >> 2, ng = task & 3;
    int m = mtile * 16 + l16, n0 = ng * 64;
    v8f acc[4];
#pragma unroll
    for (int j = 0; j < 4; ++j) acc[j] = zero8();
    for (int k0 = 0; k0 < D_; k0 += 4) {
        int kb = k0 + 2 * half;
        v2f a = *(const v2f*)(src + (size_t)m * D_ + kb);
#pragma unroll
        for (int jt = 0; jt < 4; ++jt) {
            int n = n0 + jt * 16 + l16;
            v2f bv; bv.x = B[(size_t)kb * F_ + n]; bv.y = B[(size_t)(kb + 1) * F_ + n];
            acc[jt] = wmma_f32x4(a, bv, acc[jt]);
        }
    }
#pragma unroll
    for (int jt = 0; jt < 4; ++jt)
#pragma unroll
        for (int r = 0; r < 8; ++r) {
            int mr = mtile * 16 + r + half * 8;
            C[(size_t)mr * F_ + n0 + jt * 16 + l16] = acc[jt][r];
        }
}

// ---- K3: E = T@W^T + b - L + T   (pred - target), f32 WMMA K=256 ----------
__global__ void k_err(const float* __restrict__ T, const float* __restrict__ L,
                      const float* __restrict__ W, const float* __restrict__ bias,
                      float* __restrict__ E) {
    int wid = threadIdx.x >> 5, lane = threadIdx.x & 31, l16 = lane & 15, half = lane >> 4;
    int task = blockIdx.x * 8 + wid;
    int mtile = task >> 2, ng = task & 3;
    int m = mtile * 16 + l16, n0 = ng * 64;
    v8f acc[4];
#pragma unroll
    for (int j = 0; j < 4; ++j) acc[j] = zero8();
    for (int k0 = 0; k0 < F_; k0 += 4) {
        int kb = k0 + 2 * half;
        v2f a = *(const v2f*)(T + (size_t)m * F_ + kb);
#pragma unroll
        for (int jt = 0; jt < 4; ++jt) {
            int n = n0 + jt * 16 + l16;
            v2f bv = *(const v2f*)(W + (size_t)n * F_ + kb);   // B[k,n] = W[n,k]
            acc[jt] = wmma_f32x4(a, bv, acc[jt]);
        }
    }
#pragma unroll
    for (int jt = 0; jt < 4; ++jt)
#pragma unroll
        for (int r = 0; r < 8; ++r) {
            int n = n0 + jt * 16 + l16;
            int mr = mtile * 16 + r + half * 8;
            size_t o = (size_t)mr * F_ + n;
            E[o] = acc[jt][r] + bias[n] - L[o] + T[o];
        }
}

// ---- K4: gW = (2/(N*F)) * E^T @ T   (f32 WMMA, K=2048) --------------------
__global__ void k_gw(const float* __restrict__ E, const float* __restrict__ T,
                     float* __restrict__ gW) {
    int wid = threadIdx.x >> 5, lane = threadIdx.x & 31, l16 = lane & 15, half = lane >> 4;
    int task = blockIdx.x * 8 + wid;             // 256 tiles
    int mi = task >> 4, ni = task & 15;
    int m = mi * 16 + l16, n = ni * 16 + l16;
    v8f acc = zero8();
    for (int k0 = 0; k0 < N_; k0 += 4) {
        int kb = k0 + 2 * half;
        v2f a; a.x = E[(size_t)kb * F_ + m]; a.y = E[(size_t)(kb + 1) * F_ + m];
        v2f b; b.x = T[(size_t)kb * F_ + n]; b.y = T[(size_t)(kb + 1) * F_ + n];
        acc = wmma_f32x4(a, b, acc);
    }
    const float s = 2.f / ((float)N_ * (float)F_);
#pragma unroll
    for (int r = 0; r < 8; ++r)
        gW[(size_t)(mi * 16 + r + half * 8) * F_ + n] = acc[r] * s;
}

// ---- K5: gb = (2/(N*F)) * colsum(E) ---------------------------------------
__global__ void k_gb(const float* __restrict__ E, float* __restrict__ gb) {
    __shared__ float red[256];
    int i = blockIdx.x, t = threadIdx.x;
    float s = 0.f;
    for (int r = 0; r < 8; ++r) s += E[(size_t)(t + 256 * r) * F_ + i];
    red[t] = s; __syncthreads();
    for (int off = 128; off > 0; off >>= 1) {
        if (t < off) red[t] += red[t + off];
        __syncthreads();
    }
    if (t == 0) gb[i] = red[0] * (2.f / ((float)N_ * (float)F_));
}

// ---- K6: post += (0.01/N) * sigmoid(src @ Wlr^T + blr)  (bf16 WMMA) -------
// One WG owns 128 Wlr rows, staged once in 256 KB LDS as bf16 (WGP LDS is
// 320 KB); streams all 2048 src rows (L2-resident) with 32Mx128N register
// tiles: per K-step 2 A-frags / 8 B-frags / 16 WMMAs.
__global__ void __launch_bounds__(256) k_lr(const __bf16* __restrict__ srcb,
                                            const float* __restrict__ Wlr,
                                            const float* __restrict__ blr,
                                            float* __restrict__ post) {
    __shared__ __align__(16) __bf16 Bs[128 * D_];       // 256 KB
    int jbase = blockIdx.x * 128;
    int t = threadIdx.x;
    // stage + convert Wlr block (128 rows x 1024) f32 -> bf16 in LDS
    for (int it = 0; it < 128; ++it) {
        int idx = (it * 256 + t) * 4;
        int row = idx >> 10, k = idx & (D_ - 1);
        float4 f = *(const float4*)(Wlr + (size_t)(jbase + row) * D_ + k);
        union { __bf16 h[4]; uint2 u; } pk;
        pk.h[0] = (__bf16)f.x; pk.h[1] = (__bf16)f.y;
        pk.h[2] = (__bf16)f.z; pk.h[3] = (__bf16)f.w;
        *(uint2*)(&Bs[row * D_ + k]) = pk.u;
    }
    __syncthreads();

    int wid = t >> 5, lane = t & 31, l16 = lane & 15, half = lane >> 4;
    float bl[8];
#pragma unroll
    for (int jt = 0; jt < 8; ++jt) bl[jt] = blr[jbase + jt * 16 + l16];
    float sums[8] = {0.f, 0.f, 0.f, 0.f, 0.f, 0.f, 0.f, 0.f};

    for (int mt = 0; mt < 8; ++mt) {                    // wave owns 256 rows, 32 per tile
        int row0 = wid * 256 + mt * 32;
        if (mt < 7) {                                   // prefetch next M-tile's src rows
            __builtin_prefetch((const void*)(srcb + (size_t)(row0 + 32 + l16) * D_), 0, 0);
            __builtin_prefetch((const void*)(srcb + (size_t)(row0 + 48 + l16) * D_), 0, 0);
        }
        v8f acc0[8], acc1[8];
#pragma unroll
        for (int j = 0; j < 8; ++j) { acc0[j] = zero8(); acc1[j] = zero8(); }
        for (int k0 = 0; k0 < D_; k0 += 32) {
            int kb = k0 + half * 8;                     // A 16x32 bf16 lane layout
            union { uint4 u[2]; v16bf v; } a0, a1, bv;
            const __bf16* p0 = srcb + (size_t)(row0 + l16) * D_ + kb;
            a0.u[0] = *(const uint4*)p0;  a0.u[1] = *(const uint4*)(p0 + 16);
            const __bf16* p1 = p0 + 16 * D_;
            a1.u[0] = *(const uint4*)p1;  a1.u[1] = *(const uint4*)(p1 + 16);
#pragma unroll
            for (int jt = 0; jt < 8; ++jt) {
                const __bf16* bp = &Bs[(size_t)(jt * 16 + l16) * D_ + kb];
                bv.u[0] = *(const uint4*)bp;  bv.u[1] = *(const uint4*)(bp + 16);
                acc0[jt] = wmma_bf16(a0.v, bv.v, acc0[jt]);
                acc1[jt] = wmma_bf16(a1.v, bv.v, acc1[jt]);
            }
        }
#pragma unroll
        for (int jt = 0; jt < 8; ++jt) {
            float bb = bl[jt], s = 0.f;
#pragma unroll
            for (int r = 0; r < 8; ++r) {
                s += sigmoidf(acc0[jt][r] + bb);
                s += sigmoidf(acc1[jt][r] + bb);
            }
            sums[jt] += s;
        }
    }
#pragma unroll
    for (int jt = 0; jt < 8; ++jt) {
        float tot = sums[jt] + __shfl_xor(sums[jt], 16, 32);
        if (lane < 16)
            atomicAdd(&post[jbase + jt * 16 + lane], tot * (0.01f / (float)N_));
    }
}

// ---- K7: W_upd = W - lr_mat*gW ; b_upd = b - lr_bias*gb -------------------
__global__ void k_upd(const float* __restrict__ W, const float* __restrict__ b,
                      const float* __restrict__ gW, const float* __restrict__ gb,
                      const float* __restrict__ post,
                      float* __restrict__ Wu, float* __restrict__ bu) {
    int idx = blockIdx.x * 256 + threadIdx.x;
    if (idx < FF_)      Wu[idx] = W[idx] - post[idx] * gW[idx];
    else if (idx < P_)  { int i = idx - FF_; bu[i] = b[i] - post[idx] * gb[i]; }
}

// ---- K8: z = Q @ Wu^T + bu + Q  (f32 WMMA, K=256) -------------------------
__global__ void k_z(const float* __restrict__ Q, const float* __restrict__ Wu,
                    const float* __restrict__ bu, float* __restrict__ Z) {
    int wid = threadIdx.x >> 5, lane = threadIdx.x & 31, l16 = lane & 15, half = lane >> 4;
    int task = blockIdx.x * 8 + wid;
    int mtile = task >> 2, ng = task & 3;
    int m = mtile * 16 + l16, n0 = ng * 64;
    v8f acc[4];
#pragma unroll
    for (int j = 0; j < 4; ++j) acc[j] = zero8();
    for (int k0 = 0; k0 < F_; k0 += 4) {
        int kb = k0 + 2 * half;
        v2f a = *(const v2f*)(Q + (size_t)m * F_ + kb);
#pragma unroll
        for (int jt = 0; jt < 4; ++jt) {
            int n = n0 + jt * 16 + l16;
            v2f bv = *(const v2f*)(Wu + (size_t)n * F_ + kb);
            acc[jt] = wmma_f32x4(a, bv, acc[jt]);
        }
    }
#pragma unroll
    for (int jt = 0; jt < 4; ++jt)
#pragma unroll
        for (int r = 0; r < 8; ++r) {
            int n = n0 + jt * 16 + l16;
            int mr = mtile * 16 + r + half * 8;
            size_t o = (size_t)mr * F_ + n;
            Z[o] = acc[jt][r] + bu[n] + Q[o];
        }
}

// ---- K9: out = Z @ theta_o  (f32 WMMA, K=256, Nout=1024) ------------------
__global__ void k_out(const float* __restrict__ Z, const float* __restrict__ To,
                      float* __restrict__ out) {
    int wid = threadIdx.x >> 5, lane = threadIdx.x & 31, l16 = lane & 15, half = lane >> 4;
    int task = blockIdx.x * 8 + wid;             // 2048 tasks
    int mtile = task >> 4, ng = task & 15;
    int m = mtile * 16 + l16, n0 = ng * 64;
    v8f acc[4];
#pragma unroll
    for (int j = 0; j < 4; ++j) acc[j] = zero8();
    for (int k0 = 0; k0 < F_; k0 += 4) {
        int kb = k0 + 2 * half;
        v2f a = *(const v2f*)(Z + (size_t)m * F_ + kb);
#pragma unroll
        for (int jt = 0; jt < 4; ++jt) {
            int n = n0 + jt * 16 + l16;
            v2f bv; bv.x = To[(size_t)kb * D_ + n]; bv.y = To[(size_t)(kb + 1) * D_ + n];
            acc[jt] = wmma_f32x4(a, bv, acc[jt]);
        }
    }
#pragma unroll
    for (int jt = 0; jt < 4; ++jt)
#pragma unroll
        for (int r = 0; r < 8; ++r) {
            int mr = mtile * 16 + r + half * 8;
            out[(size_t)mr * D_ + n0 + jt * 16 + l16] = acc[jt][r];
        }
}

// ---- host-side launcher ----------------------------------------------------
extern "C" void kernel_launch(void* const* d_in, const int* in_sizes, int n_in,
                              void* d_out, int out_size, void* d_ws, size_t ws_size,
                              hipStream_t stream) {
    (void)in_sizes; (void)n_in; (void)out_size; (void)ws_size;
    const float* src = (const float*)d_in[0];
    const float* tk  = (const float*)d_in[1];
    const float* tq  = (const float*)d_in[2];
    const float* tv  = (const float*)d_in[3];
    const float* to  = (const float*)d_in[4];
    const float* W   = (const float*)d_in[5];
    const float* b   = (const float*)d_in[6];
    const float* Wlr = (const float*)d_in[7];
    const float* blr = (const float*)d_in[8];
    float* out = (float*)d_out;

    float* ws = (float*)d_ws;
    size_t o = 0;
    float* Tv   = ws + o; o += (size_t)N_ * F_;
    float* Lv   = ws + o; o += (size_t)N_ * F_;
    float* Qv   = ws + o; o += (size_t)N_ * F_;
    float* E    = ws + o; o += (size_t)N_ * F_;
    float* Zv   = ws + o; o += (size_t)N_ * F_;
    float* gWp  = ws + o; o += FF_;
    float* gbp  = ws + o; o += F_;
    float* post = ws + o; o += P_;
    float* Wu   = ws + o; o += FF_;
    float* bu   = ws + o; o += F_;
    __bf16* srcb = (__bf16*)(ws + o);            // 4 MB bf16 copy of src

    k_prep<<<2048, 256, 0, stream>>>(src, srcb, post);
    k_proj<<<dim3(64, 3), 256, 0, stream>>>(src, tk, tq, tv, Tv, Qv, Lv);
    k_err <<<64,   256, 0, stream>>>(Tv, Lv, W, b, E);
    k_gw  <<<32,   256, 0, stream>>>(E, Tv, gWp);
    k_gb  <<<256,  256, 0, stream>>>(E, gbp);
    k_lr  <<<514,  256, 0, stream>>>(srcb, Wlr, blr, post);
    k_upd <<<257,  256, 0, stream>>>(W, b, gWp, gbp, post, Wu, bu);
    k_z   <<<64,   256, 0, stream>>>(Qv, Wu, bu, Zv);
    k_out <<<256,  256, 0, stream>>>(Zv, to, out);
}
